// Llama4VisionAttention_37151467110432
// MI455X (gfx1250) — compile-verified, matrix operand order checked by reading, not run on hardware
//
#include <hip/hip_runtime.h>

// ---------------------------------------------------------------------------
// Llama4 Vision Attention forward on gfx1250 (MI455X), wave32 + WMMA bf16.
//   B=16, S=577, D=1408, H=16, HD=88.  HD padded to 96, S padded to 640.
// Matrix math: v_wmma_f32_16x16x32_bf16.  GEMM tiles move via
// global_load_async_to_lds_b128 (ASYNCcnt) with LDS double buffering.
// ---------------------------------------------------------------------------

typedef __attribute__((ext_vector_type(16))) __bf16 v16bf;
typedef __attribute__((ext_vector_type(8)))  __bf16 v8bf;
typedef __attribute__((ext_vector_type(8)))  float  v8f;

#define NB 16
#define NS 577
#define ND 1408
#define NH 16
#define HD 88
#define HDP 96            // padded head dim (3 x K=32 WMMA steps)
#define SP  640           // padded sequence (10 x 64-query tiles, >= 19*32 keys)
#define M_ROWS (NB*NS)    // 9232
#define N_QKV  (3*ND)     // 4224
#define ATT_SCALE 0.10660036f   // 1/sqrt(88)

// ---- WMMA helpers ---------------------------------------------------------

static __device__ inline v8f wmma_bf16(v16bf a, v16bf b, v8f c) {
  // 8 args: (neg_a, A, neg_b, B, c_mod, C, reuse_a, reuse_b)
  return __builtin_amdgcn_wmma_f32_16x16x32_bf16(false, a, false, b,
                                                 (short)0, c, false, false);
}

// A/B 16x32 bf16 fragment: lane half h picks K-chunks {8h..8h+7, 16+8h..16+8h+7}
static __device__ inline v16bf load_frag(const __bf16* p, int h) {
  const v8bf* p0 = (const v8bf*)__builtin_assume_aligned(p + 8*h, 16);
  const v8bf* p1 = (const v8bf*)__builtin_assume_aligned(p + 16 + 8*h, 16);
  v8bf lo = *p0, hi = *p1;
  v16bf f;
#pragma unroll
  for (int i = 0; i < 8; ++i) { f[i] = lo[i]; f[i + 8] = hi[i]; }
  return f;
}

// async 16B global -> LDS copy, tracked with ASYNCcnt (CDNA5)
static __device__ inline void async_copy_b128(unsigned lds_off, const void* gaddr) {
  asm volatile("global_load_async_to_lds_b128 %0, %1, off"
               :: "v"(lds_off), "v"(gaddr) : "memory");
}
static __device__ inline void wait_async0() {
  asm volatile("s_wait_asynccnt 0x0" ::: "memory");
}
// flat shared pointer -> raw LDS byte offset (low 32 bits of LDS aperture addr)
static __device__ inline unsigned lds_off32(const void* p) {
  return (unsigned)(unsigned long long)p;
}

// ---- fp32 -> bf16 convert -------------------------------------------------

__global__ __launch_bounds__(256)
void cvt_f32_bf16(const float* __restrict__ src, __bf16* __restrict__ dst, long n) {
  long i = (long)blockIdx.x * blockDim.x + threadIdx.x;
  long stride = (long)gridDim.x * blockDim.x;
  for (; i < n; i += stride) dst[i] = (__bf16)src[i];
}

// ---- Tiled WMMA GEMM: C[M,N] = A[M,K] * Bt[N,K]^T + bias ------------------
// Block tile 128x128, 8 waves (4m x 2n), each wave 32x64 (2x4 WMMA tiles).
// K-step 32, LDS double-buffered, filled by async global->LDS b128 copies.
// LDS row pitch 56 bf16 (112B): 16B-aligned rows, conflict-free b128 reads.

#define LDS_PITCH 56
#define TILE_ELEMS (128 * LDS_PITCH)

__global__ __launch_bounds__(256)
void gemm_bf16_wmma(const __bf16* __restrict__ A, const __bf16* __restrict__ Bt,
                    const float* __restrict__ bias, void* __restrict__ Cout,
                    int M, int N, int K, int out_f32) {
  __shared__ __align__(16) __bf16 ldsA[2 * TILE_ELEMS];
  __shared__ __align__(16) __bf16 ldsB[2 * TILE_ELEMS];

  const int tid  = threadIdx.x;
  const int wave = tid >> 5, lane = tid & 31;
  const int h = lane >> 4, idx = lane & 15;
  const int wm = wave & 3, wn = wave >> 2;          // 4x2 wave grid
  const int m0 = blockIdx.x * 128, n0 = blockIdx.y * 128;

  // this thread's two 16B chunks per tile: rows (tid>>2) and (tid>>2)+64
  const int row0 = tid >> 2, col8 = (tid & 3) * 8;
  // A rows clamped to M-1: garbage rows only feed garbage C rows (not stored)
  const int ga0 = (m0 + row0      < M) ? m0 + row0      : M - 1;
  const int ga1 = (m0 + row0 + 64 < M) ? m0 + row0 + 64 : M - 1;
  const int gb0 = n0 + row0, gb1 = n0 + row0 + 64;   // N multiple of 128

  v8f acc[2][4];
#pragma unroll
  for (int i = 0; i < 2; ++i)
#pragma unroll
    for (int j = 0; j < 4; ++j)
#pragma unroll
      for (int r = 0; r < 8; ++r) acc[i][j][r] = 0.f;

  const int nk = K >> 5;   // K/32, exact (1408/32 = 44)

  // prologue: async-fill buffer 0 with tile kt=0
  {
    unsigned la0 = lds_off32(&ldsA[row0 * LDS_PITCH + col8]);
    unsigned lb0 = lds_off32(&ldsB[row0 * LDS_PITCH + col8]);
    async_copy_b128(la0,                            A  + (long)ga0 * K + col8);
    async_copy_b128(la0 + 64u * LDS_PITCH * 2u,     A  + (long)ga1 * K + col8);
    async_copy_b128(lb0,                            Bt + (long)gb0 * K + col8);
    async_copy_b128(lb0 + 64u * LDS_PITCH * 2u,     Bt + (long)gb1 * K + col8);
  }
  wait_async0();
  __syncthreads();

  for (int t = 0; t < nk; ++t) {
    const int buf = t & 1;
    // software pipeline: kick off next tile into the other buffer
    if (t + 1 < nk) {
      const int kt = (t + 1) << 5;
      unsigned la = lds_off32(&ldsA[(buf ^ 1) * TILE_ELEMS + row0 * LDS_PITCH + col8]);
      unsigned lb = lds_off32(&ldsB[(buf ^ 1) * TILE_ELEMS + row0 * LDS_PITCH + col8]);
      async_copy_b128(la,                          A  + (long)ga0 * K + kt + col8);
      async_copy_b128(la + 64u * LDS_PITCH * 2u,   A  + (long)ga1 * K + kt + col8);
      async_copy_b128(lb,                          Bt + (long)gb0 * K + kt + col8);
      async_copy_b128(lb + 64u * LDS_PITCH * 2u,   Bt + (long)gb1 * K + kt + col8);
    }
    // consume current buffer: 2 A-frags x 4 B-frags -> 8 WMMAs
    const __bf16* aB = &ldsA[buf * TILE_ELEMS];
    const __bf16* bB = &ldsB[buf * TILE_ELEMS];
    v16bf af[2], bf4[4];
#pragma unroll
    for (int sm = 0; sm < 2; ++sm)
      af[sm] = load_frag(&aB[(wm * 32 + sm * 16 + idx) * LDS_PITCH], h);
#pragma unroll
    for (int sn = 0; sn < 4; ++sn)
      bf4[sn] = load_frag(&bB[(wn * 64 + sn * 16 + idx) * LDS_PITCH], h);
#pragma unroll
    for (int sm = 0; sm < 2; ++sm)
#pragma unroll
      for (int sn = 0; sn < 4; ++sn)
        acc[sm][sn] = wmma_bf16(af[sm], bf4[sn], acc[sm][sn]);

    wait_async0();       // own async writes to buf^1 have landed in LDS
    __syncthreads();     // everyone's writes visible, everyone done reading buf
  }

  // epilogue: C layout row = r + 8h, col = idx within each 16x16 tile
#pragma unroll
  for (int sm = 0; sm < 2; ++sm)
#pragma unroll
    for (int sn = 0; sn < 4; ++sn) {
      int col = n0 + wn * 64 + sn * 16 + idx;
      float bv = bias[col];
#pragma unroll
      for (int r = 0; r < 8; ++r) {
        int row = m0 + wm * 32 + sm * 16 + r + 8 * h;
        if (row < M) {
          float v = acc[sm][sn][r] + bv;
          if (out_f32) ((float*)Cout)[(long)row * N + col] = v;
          else         ((__bf16*)Cout)[(long)row * N + col] = (__bf16)v;
        }
      }
    }
}

// ---- RoPE + scatter into padded Q/K [BH][SP][HDP] and V^T [BH][HDP][SP] ---

__global__ __launch_bounds__(256)
void rope_scatter(const __bf16* __restrict__ qkv, const float* __restrict__ freqs,
                  __bf16* __restrict__ Qp, __bf16* __restrict__ Kp,
                  __bf16* __restrict__ Vt) {
  long t = (long)blockIdx.x * 256 + threadIdx.x;   // B*H*SP*48 threads exactly
  int j = (int)(t % 48); long r = t / 48;
  int s = (int)(r % SP); r /= SP;
  int h = (int)(r % NH); int b = (int)(r / NH);
  int bh = b * NH + h;

  float qe = 0.f, qo = 0.f, ke = 0.f, ko = 0.f, v0 = 0.f, v1 = 0.f;
  if (j < (HD / 2) && s < NS) {
    long base = ((long)(b * NS + s)) * N_QKV + h * HD + 2 * j;
    float q0 = (float)qkv[base],          q1 = (float)qkv[base + 1];
    float k0 = (float)qkv[base + ND],     k1 = (float)qkv[base + ND + 1];
    v0 = (float)qkv[base + 2 * ND];       v1 = (float)qkv[base + 2 * ND + 1];
    float cs = freqs[((long)s * (HD / 2) + j) * 2 + 0];
    float sn = freqs[((long)s * (HD / 2) + j) * 2 + 1];
    qe = q0 * cs - q1 * sn;  qo = q0 * sn + q1 * cs;
    ke = k0 * cs - k1 * sn;  ko = k0 * sn + k1 * cs;
  }
  long qbase = ((long)bh * SP + s) * HDP + 2 * j;
  Qp[qbase] = (__bf16)qe;  Qp[qbase + 1] = (__bf16)qo;
  Kp[qbase] = (__bf16)ke;  Kp[qbase + 1] = (__bf16)ko;
  Vt[((long)bh * HDP + 2 * j) * SP + s]     = (__bf16)v0;
  Vt[((long)bh * HDP + 2 * j + 1) * SP + s] = (__bf16)v1;
}

// ---- Flash attention: 4 waves x 16 queries, 32 keys/step, all WMMA --------

#define P_PITCH 56

__global__ __launch_bounds__(128)
void attn_flash_wmma(const __bf16* __restrict__ Qp, const __bf16* __restrict__ Kp,
                     const __bf16* __restrict__ Vt, __bf16* __restrict__ Ctx) {
  __shared__ __align__(16) __bf16 ldsP[4 * 16 * P_PITCH];
  const int tid = threadIdx.x;
  const int w = tid >> 5, lane = tid & 31;
  const int h = lane >> 4, idx = lane & 15;
  const int qtile = blockIdx.x;            // 0..9
  const int bh = blockIdx.y;               // 0..255
  const int q0 = qtile * 64 + w * 16;
  __bf16* myP = ldsP + w * 16 * P_PITCH;

  // Q fragments (loop invariant): 16 rows x 96 K, 3 x v16bf
  const __bf16* Qrow = Qp + ((long)bh * SP + q0 + idx) * HDP;
  v16bf qf[3];
#pragma unroll
  for (int kk = 0; kk < 3; ++kk) qf[kk] = load_frag(Qrow + kk * 32, h);

  v8f ctx[6];
  float m[8], l[8];
#pragma unroll
  for (int t = 0; t < 6; ++t)
#pragma unroll
    for (int r = 0; r < 8; ++r) ctx[t][r] = 0.f;
#pragma unroll
  for (int r = 0; r < 8; ++r) { m[r] = -1e30f; l[r] = 0.f; }

  for (int k0 = 0; k0 < NS; k0 += 32) {
    // S = Q K^T for 32 keys (two 16-key N tiles), K-dim 96 = 3 WMMAs each
    v8f sc[2];
#pragma unroll
    for (int sub = 0; sub < 2; ++sub) {
#pragma unroll
      for (int r = 0; r < 8; ++r) sc[sub][r] = 0.f;
      const __bf16* Krow = Kp + ((long)bh * SP + k0 + sub * 16 + idx) * HDP;
#pragma unroll
      for (int kk = 0; kk < 3; ++kk) {
        v16bf kf = load_frag(Krow + kk * 32, h);
        sc[sub] = wmma_bf16(qf[kk], kf, sc[sub]);
      }
    }
    // scale + key mask (col = k0 + sub*16 + idx, independent of r)
    const bool ok0 = (k0 + idx) < NS;
    const bool ok1 = (k0 + 16 + idx) < NS;
    float mx[8];
#pragma unroll
    for (int r = 0; r < 8; ++r) {
      float s0 = ok0 ? sc[0][r] * ATT_SCALE : -1e30f;
      float s1 = ok1 ? sc[1][r] * ATT_SCALE : -1e30f;
      sc[0][r] = s0; sc[1][r] = s1;
      mx[r] = fmaxf(s0, s1);
    }
    // row max across the 16 lanes of each half (C layout: row = r + 8h)
#pragma unroll
    for (int d = 1; d < 16; d <<= 1)
#pragma unroll
      for (int r = 0; r < 8; ++r) mx[r] = fmaxf(mx[r], __shfl_xor(mx[r], d, 32));

    float alpha[8], rs[8];
#pragma unroll
    for (int r = 0; r < 8; ++r) {
      float mn = fmaxf(m[r], mx[r]);
      alpha[r] = __expf(m[r] - mn);
      m[r] = mn;
      float p0 = __expf(sc[0][r] - mn);
      float p1 = __expf(sc[1][r] - mn);
      // stage P (C layout) into LDS for re-layout as an A fragment
      myP[(r + 8 * h) * P_PITCH + idx]      = (__bf16)p0;
      myP[(r + 8 * h) * P_PITCH + 16 + idx] = (__bf16)p1;
      rs[r] = p0 + p1;
    }
#pragma unroll
    for (int d = 1; d < 16; d <<= 1)
#pragma unroll
      for (int r = 0; r < 8; ++r) rs[r] += __shfl_xor(rs[r], d, 32);
#pragma unroll
    for (int r = 0; r < 8; ++r) l[r] = l[r] * alpha[r] + rs[r];
#pragma unroll
    for (int t = 0; t < 6; ++t)
#pragma unroll
      for (int r = 0; r < 8; ++r) ctx[t][r] *= alpha[r];

    __builtin_amdgcn_wave_barrier();       // order in-wave LDS write -> read
    v16bf pf = load_frag(myP + idx * P_PITCH, h);
    __builtin_amdgcn_wave_barrier();       // reads before next-iter writes

    // ctx += P (16x32) @ V (32x96); Vt rows are d, contiguous in key
#pragma unroll
    for (int t = 0; t < 6; ++t) {
      const __bf16* Vrow = Vt + ((long)bh * HDP + t * 16 + idx) * SP + k0;
      v16bf vf = load_frag(Vrow, h);
      ctx[t] = wmma_bf16(pf, vf, ctx[t]);
    }
  }

  // normalize and scatter into ctx buffer [B*S][D] (bf16), drop the d>=88 pad
  const int b = bh >> 4, head = bh & 15;
#pragma unroll
  for (int r = 0; r < 8; ++r) {
    int srow = q0 + r + 8 * h;
    if (srow < NS) {
      float inv = 1.f / l[r];
      long base = ((long)(b * NS + srow)) * ND + head * HD;
#pragma unroll
      for (int t = 0; t < 6; ++t) {
        int d = t * 16 + idx;
        if (d < HD) Ctx[base + d] = (__bf16)(ctx[t][r] * inv);
      }
    }
  }
}

// ---- host launcher --------------------------------------------------------

extern "C" void kernel_launch(void* const* d_in, const int* in_sizes, int n_in,
                              void* d_out, int out_size, void* d_ws, size_t ws_size,
                              hipStream_t stream) {
  (void)in_sizes; (void)n_in; (void)out_size; (void)ws_size;
  const float* hidden = (const float*)d_in[0];
  const float* freqs  = (const float*)d_in[1];
  const float* wqkv   = (const float*)d_in[2];
  const float* bqkv   = (const float*)d_in[3];
  const float* wo     = (const float*)d_in[4];
  const float* bo     = (const float*)d_in[5];

  char* ws = (char*)d_ws;
  size_t off = 0;
  auto take = [&](size_t bytes) -> char* {
    char* p = ws + off;
    off += (bytes + 255) & ~(size_t)255;
    return p;
  };
  __bf16* Xb    = (__bf16*)take((size_t)M_ROWS * ND * 2);
  __bf16* Wqkvb = (__bf16*)take((size_t)N_QKV * ND * 2);
  __bf16* Wob   = (__bf16*)take((size_t)ND * ND * 2);
  __bf16* QKVb  = (__bf16*)take((size_t)M_ROWS * N_QKV * 2);
  __bf16* Qp    = (__bf16*)take((size_t)NB * NH * SP * HDP * 2);
  __bf16* Kp    = (__bf16*)take((size_t)NB * NH * SP * HDP * 2);
  __bf16* Vt    = (__bf16*)take((size_t)NB * NH * HDP * SP * 2);
  __bf16* Ctxb  = (__bf16*)take((size_t)M_ROWS * ND * 2);

  // 1) fp32 -> bf16
  cvt_f32_bf16<<<4096, 256, 0, stream>>>(hidden, Xb, (long)M_ROWS * ND);
  cvt_f32_bf16<<<4096, 256, 0, stream>>>(wqkv, Wqkvb, (long)N_QKV * ND);
  cvt_f32_bf16<<<2048, 256, 0, stream>>>(wo, Wob, (long)ND * ND);

  // 2) QKV = X @ Wqkv^T + b  (bf16 out);  N tiles of 128: 4224/128 = 33
  dim3 g1((M_ROWS + 127) / 128, N_QKV / 128);
  gemm_bf16_wmma<<<g1, 256, 0, stream>>>(Xb, Wqkvb, bqkv, (void*)QKVb,
                                         M_ROWS, N_QKV, ND, 0);

  // 3) RoPE + scatter to padded Q/K/V^T
  long rope_threads = (long)NB * NH * SP * 48;          // 7,864,320
  rope_scatter<<<(unsigned)(rope_threads / 256), 256, 0, stream>>>(
      QKVb, freqs, Qp, Kp, Vt);

  // 4) flash attention (10 q-tiles x 256 (b,h) pairs)
  attn_flash_wmma<<<dim3(SP / 64, NB * NH), 128, 0, stream>>>(Qp, Kp, Vt, Ctxb);

  // 5) out = ctx @ Wo^T + bo  (fp32 into d_out);  1408/128 = 11
  dim3 g2((M_ROWS + 127) / 128, ND / 128);
  gemm_bf16_wmma<<<g2, 256, 0, stream>>>(Ctxb, Wob, bo, d_out,
                                         M_ROWS, ND, ND, 1);
}
